// RPN_76038101009034
// MI455X (gfx1250) — matrix operand order repaired and checked by reading, not compiled, not run on hardware
//
#include <hip/hip_runtime.h>

// ---------------------------------------------------------------------------
// RPN for MI455X (gfx1250, wave32, WMMA).
// conv1 3x3 512->512 runs as implicit GEMM on v_wmma_f32_16x16x32_bf16,
// double-buffered LDS staging, immediate-offset A-fragment loads.
// ---------------------------------------------------------------------------

typedef __attribute__((ext_vector_type(16))) __bf16 v16bf;
typedef __attribute__((ext_vector_type(8)))  float  v8f;

#define NEGV (-1e30f)

// ---- sizes -----------------------------------------------------------------
#define BATCH   8
#define CIN     512
#define COUT    512
#define HH      64
#define WW      64
#define NPOS    (HH*WW)              // 4096
#define NANCH   9
#define NBOX    (NPOS*NANCH)         // 36864
#define PRE_NMS 6000
#define POST_NMS 300

// ---- workspace layout (bytes) ---------------------------------------------
#define OFF_XBF    ((size_t)0)                      // 8*512*64*64 bf16 = 33,554,432
#define OFF_WFRAG  ((size_t)33554432)               // 512*4608 bf16    =  4,718,592
#define OFF_FEATS  ((size_t)38273024)               // NHWC bf16        = 33,554,432
#define OFF_FG     ((size_t)71827456)               // 8*36864 f32      =  1,179,648
#define OFF_BOXES  ((size_t)73007104)               // 8*36864*4 f32    =  4,718,592
#define OFF_SKEY   ((size_t)77725696)               // 8*36864 f32      =  1,179,648
#define OFF_CBOX   ((size_t)78905344)               // 8*6000*4 f32     =    768,000
#define OFF_CSC    ((size_t)79673344)               // 8*6000 f32       =    192,000

// ---- output layout (float elements) ---------------------------------------
#define O_OBJ   ((size_t)0)                         // 8*36864*2 = 589824
#define O_LOC   ((size_t)589824)                    // 8*36864*4 = 1179648
#define O_ROIS  ((size_t)1769472)                   // 8*300*4   = 9600
#define O_RIDX  ((size_t)1779072)                   // 8*300     = 2400
#define O_ANCH  ((size_t)1781472)                   // 36864*4   = 147456

// ===========================================================================
// f32 -> bf16 activation convert (one streaming pass; x then lives in L2)
// ===========================================================================
__global__ void k_cvt_x(const float* __restrict__ x, __bf16* __restrict__ xbf, int n) {
  int i = blockIdx.x * blockDim.x + threadIdx.x;
  if (i < n) xbf[i] = (__bf16)x[i];
}

// ===========================================================================
// Reformat conv1 weights into WMMA A-fragment register order (bf16).
// Tile t = (kt*16 + ct)*9 + p holds a 16(M=kout) x 32(K=cin-chunk) matrix at
// fixed tap p = dy*3+dx, stored lane-major, contiguous K order (K = 16*half + j), same as B:
//   K in 0..31: half=K/16; j=K%16 -> v=j/2, hw=j&1
//   element = lane*16 + v*2 + hw,  lane = half*16 + M
// ===========================================================================
__global__ void k_reformat_w(const float* __restrict__ w, __bf16* __restrict__ wf, int n) {
  int i = blockIdx.x * blockDim.x + threadIdx.x;
  if (i >= n) return;
  int dx = i % 3; int t1 = i / 3;
  int dy = t1 % 3; int t2 = t1 / 3;
  int c  = t2 % CIN; int k = t2 / CIN;
  int kt = k >> 4, m = k & 15, ct = c >> 5, K = c & 31, p = dy * 3 + dx;
  int half = K >> 4, j = K & 15, v = j >> 1, hw = j & 1;
  int tile = (kt * 16 + ct) * 9 + p;
  wf[(size_t)tile * 512 + (size_t)(half * 16 + m) * 16 + v * 2 + hw] = (__bf16)w[i];
}

// ===========================================================================
// conv1: 3x3, pad 1, implicit GEMM with v_wmma_f32_16x16x32_bf16.
// grid = (row y:64, kblock:8, batch:8), block = 256 (8 wave32).
// Wave (ksub = wave&3, xpair = wave>>2) owns 16 out-channels x 32 pixels
// (two 16x16 accumulators). K loop: 16 c-chunks x 9 taps = 288 WMMA / wave.
// Double-buffered LDS staging: chunk ct+1 loads drain behind chunk ct WMMAs.
// ===========================================================================
#define CSTR 48   // LDS channel stride (32 used + pad): keeps 32B alignment, low conflicts
#define XTILE (3 * 66 * CSTR)

__global__ __launch_bounds__(256)
void k_conv1(const __bf16* __restrict__ xbf, const __bf16* __restrict__ wf,
             const float* __restrict__ bias, __bf16* __restrict__ feats) {
  __shared__ __attribute__((aligned(32))) __bf16 xlds[2][XTILE];

  const int tid  = threadIdx.x;
  const int lane = tid & 31, wave = tid >> 5;
  const int ksub = wave & 3, xpair = wave >> 2;
  const int y  = blockIdx.x, kb = blockIdx.y, n = blockIdx.z;
  const int halfId = lane >> 4, lmod = lane & 15;

  // C/D layout: lane, vgpr r -> M = halfId*8 + r, N = lmod. Init with bias.
  const int kbase = kb * 64 + ksub * 16 + halfId * 8;
  v8f acc0, acc1;
  #pragma unroll
  for (int r = 0; r < 8; ++r) { float bv = bias[kbase + r]; acc0[r] = bv; acc1[r] = bv; }

  const int kt = kb * 4 + ksub;

  // stage 32 channels x 3 rows x 66 px (halo, zero-padded) into LDS buffer
  auto stage = [&](int ct, int buf) {
    const int c0 = ct * 32;
    for (int idx = tid; idx < 32 * 3 * 66; idx += 256) {
      int c   = idx / 198;
      int rem = idx - c * 198;
      int rr  = rem / 66;
      int xs  = rem - rr * 66;
      int yi  = y + rr - 1, xi = xs - 1;
      __bf16 v = (__bf16)0.0f;
      if ((unsigned)yi < 64u && (unsigned)xi < 64u)
        v = xbf[(((size_t)n * CIN + c0 + c) * HH + yi) * WW + xi];
      xlds[buf][(rr * 66 + xs) * CSTR + c] = v;
    }
  };

  stage(0, 0);
  __syncthreads();

  for (int ct = 0; ct < 16; ++ct) {
    const int buf = ct & 1;
    if (ct + 1 < 16) {
      stage(ct + 1, buf ^ 1);                 // overlaps with WMMAs below
      if (ct + 2 < 16)                        // gfx1250 global_prefetch_b8
        __builtin_prefetch(
            (const void*)(xbf + (((size_t)n * CIN + (ct + 2) * 32 + lane) * HH + y) * WW),
            0, 1);
    }

    // A: uniform base per chunk -> all 9 taps are immediate-offset b128 loads
    const __bf16* wbase = wf + ((size_t)(kt * 16 + ct) * 9) * 512 + (size_t)lane * 16;
    // B: per-lane base; tap offsets fold into ds_load immediates
    const __bf16* bbase = &xlds[buf][(xpair * 32 + lmod) * CSTR + halfId * 16];

    #pragma unroll
    for (int p = 0; p < 9; ++p) {
      const int rr = p / 3, dxx = p - rr * 3;
      v16bf a = *(const v16bf*)(wbase + p * 512);
      const v16bf b0 = *(const v16bf*)(bbase + (rr * 66 + dxx) * CSTR);
      const v16bf b1 = *(const v16bf*)(bbase + (rr * 66 + dxx + 16) * CSTR);
      acc0 = __builtin_amdgcn_wmma_f32_16x16x32_bf16(false, a, false, b0, (short)0, acc0, false, false);
      acc1 = __builtin_amdgcn_wmma_f32_16x16x32_bf16(false, a, false, b1, (short)0, acc1, false, false);
    }
    __syncthreads();
  }

  // -- epilogue: feats NHWC bf16, packed pair stores --
  #pragma unroll
  for (int tI = 0; tI < 2; ++tI) {
    v8f acc = tI ? acc1 : acc0;
    int x = xpair * 32 + tI * 16 + lmod;
    size_t base = (((size_t)n * NPOS + y * WW + x) * COUT + kbase);
    #pragma unroll
    for (int r = 0; r < 8; r += 2) {
      union { unsigned u; __bf16 h[2]; } pk;
      pk.h[0] = (__bf16)acc[r];
      pk.h[1] = (__bf16)acc[r + 1];
      *(unsigned*)(feats + base + r) = pk.u;
    }
  }
}

// ===========================================================================
// 1x1 heads (obj 18ch + loc 36ch) + softmax fg.  One WG per (n,row).
// All 54 filters staged in LDS as bf16 (55 KB). Wave w handles 8 pixels.
// ===========================================================================
__global__ __launch_bounds__(256)
void k_heads(const __bf16* __restrict__ feats,
             const float* __restrict__ objw, const float* __restrict__ objb,
             const float* __restrict__ locw, const float* __restrict__ locb,
             float* __restrict__ out_obj, float* __restrict__ out_loc,
             float* __restrict__ fg) {
  __shared__ __attribute__((aligned(32))) __bf16 hw[54 * 512];
  const int tid = threadIdx.x;
  for (int i = tid; i < 54 * 512; i += 256) {
    int o = i >> 9, c = i & 511;
    float v = (o < 18) ? objw[o * 512 + c] : locw[(o - 18) * 512 + c];
    hw[i] = (__bf16)v;
  }
  __syncthreads();

  const int n = blockIdx.x >> 6, y = blockIdx.x & 63;
  const int wave = tid >> 5, lane = tid & 31;

  for (int xq = 0; xq < 8; ++xq) {
    const int x = wave * 8 + xq;
    const int pos = y * WW + x;
    const v16bf f = *(const v16bf*)(feats + (((size_t)n * NPOS + pos) * COUT + lane * 16));
    float fl[16];
    #pragma unroll
    for (int j = 0; j < 16; ++j) fl[j] = (float)f[j];

    float res[54];
    for (int o = 0; o < 54; ++o) {
      const v16bf wv = *(const v16bf*)(&hw[o * 512 + lane * 16]);
      float s = 0.f;
      #pragma unroll
      for (int j = 0; j < 16; ++j) s += fl[j] * (float)wv[j];
      #pragma unroll
      for (int m = 16; m; m >>= 1) s += __shfl_xor(s, m);
      res[o] = s;
    }
    if (lane == 0) {
      const size_t ob = ((size_t)n * NPOS + pos) * 18;
      const size_t lb = ((size_t)n * NPOS + pos) * 36;
      for (int a = 0; a < 9; ++a) {
        float o0 = res[2 * a]     + objb[2 * a];
        float o1 = res[2 * a + 1] + objb[2 * a + 1];
        out_obj[ob + 2 * a]     = o0;
        out_obj[ob + 2 * a + 1] = o1;
        fg[(size_t)n * NBOX + pos * 9 + a] = 1.f / (1.f + expf(o0 - o1));
      }
      for (int c = 0; c < 36; ++c) out_loc[lb + c] = res[18 + c] + locb[c];
    }
  }
}

// ===========================================================================
// anchors
// ===========================================================================
__global__ void k_anchors(float* __restrict__ anch) {
  int i = blockIdx.x * blockDim.x + threadIdx.x;
  if (i >= NBOX) return;
  int a = i % 9, pos = i / 9;
  int xx = pos % WW, yy = pos / WW;
  const float ratios[3] = {0.5f, 1.0f, 2.0f};
  const float scales[3] = {8.0f, 16.0f, 32.0f};
  float r = ratios[a / 3], s = scales[a % 3];
  float hs = 16.f * s * sqrtf(r);
  float ws = 16.f * s * sqrtf(1.f / r);
  float cy = yy * 16.f + 8.f, cx = xx * 16.f + 8.f;
  anch[(size_t)i * 4 + 0] = cy - 0.5f * hs;
  anch[(size_t)i * 4 + 1] = cx - 0.5f * ws;
  anch[(size_t)i * 4 + 2] = cy + 0.5f * hs;
  anch[(size_t)i * 4 + 3] = cx + 0.5f * ws;
}

// ===========================================================================
// decode + clip + min-size mask -> boxes, masked scores
// ===========================================================================
__global__ void k_decode(const float* __restrict__ anch, const float* __restrict__ loc,
                         const float* __restrict__ fg,
                         const int* __restrict__ ihp, const int* __restrict__ iwp,
                         float* __restrict__ boxes, float* __restrict__ skey) {
  int gid = blockIdx.x * blockDim.x + threadIdx.x;
  if (gid >= BATCH * NBOX) return;
  const float Hc = (float)ihp[0], Wc = (float)iwp[0];
  int b = gid / NBOX, i = gid - b * NBOX;
  const float* A = anch + (size_t)i * 4;
  float a0 = A[0], a1 = A[1], a2 = A[2], a3 = A[3];
  float ah = a2 - a0, aw = a3 - a1;
  float ay = a0 + 0.5f * ah, ax = a1 + 0.5f * aw;
  const float* L = loc + ((size_t)b * NBOX + i) * 4;
  float dy = L[0], dx = L[1], dh = L[2], dw = L[3];
  float cy = dy * ah + ay, cx = dx * aw + ax;
  float hh = expf(dh) * ah, wv = expf(dw) * aw;
  float y1 = fminf(fmaxf(cy - 0.5f * hh, 0.f), Hc);
  float x1 = fminf(fmaxf(cx - 0.5f * wv, 0.f), Wc);
  float y2 = fminf(fmaxf(cy + 0.5f * hh, 0.f), Hc);
  float x2 = fminf(fmaxf(cx + 0.5f * wv, 0.f), Wc);
  float* B = boxes + (size_t)gid * 4;
  B[0] = y1; B[1] = x1; B[2] = y2; B[3] = x2;
  bool valid = ((y2 - y1) >= 16.f) && ((x2 - x1) >= 16.f);
  skey[gid] = valid ? fg[gid] : NEGV;
}

// ===========================================================================
// top-6000 per batch: 2-level radix select (16-bit prefix) + compaction
// ===========================================================================
__device__ __forceinline__ unsigned okey(float f) {
  unsigned u = __float_as_uint(f);
  return (u & 0x80000000u) ? ~u : (u | 0x80000000u);
}

__global__ __launch_bounds__(1024)
void k_topk(const float* __restrict__ skey, const float* __restrict__ boxes,
            float* __restrict__ cbox, float* __restrict__ csc) {
  __shared__ unsigned hist[256], hist2[256];
  __shared__ unsigned sPivot1, sG1, sPivot16, sG, sR, cnt0, cnt1;
  const int b = blockIdx.x, tid = threadIdx.x;
  if (tid < 256) { hist[tid] = 0; hist2[tid] = 0; }
  __syncthreads();

  for (int i = tid; i < NBOX; i += 1024)
    atomicAdd(&hist[okey(skey[(size_t)b * NBOX + i]) >> 24], 1u);
  __syncthreads();
  if (tid == 0) {
    unsigned acc = 0; int pb = 0;
    for (int d = 255; d >= 0; --d) {
      unsigned h = hist[d];
      if (acc + h >= (unsigned)PRE_NMS) { pb = d; break; }
      acc += h;
    }
    sPivot1 = (unsigned)pb; sG1 = acc;
  }
  __syncthreads();

  for (int i = tid; i < NBOX; i += 1024) {
    unsigned key = okey(skey[(size_t)b * NBOX + i]);
    if ((key >> 24) == sPivot1) atomicAdd(&hist2[(key >> 16) & 255u], 1u);
  }
  __syncthreads();
  if (tid == 0) {
    unsigned acc = sG1; int pb = 0;
    for (int d = 255; d >= 0; --d) {
      unsigned h = hist2[d];
      if (acc + h >= (unsigned)PRE_NMS) { pb = d; break; }
      acc += h;
    }
    sPivot16 = (sPivot1 << 8) | (unsigned)pb;
    sG = acc; sR = (unsigned)PRE_NMS - acc;
    cnt0 = 0; cnt1 = 0;
  }
  __syncthreads();

  for (int i = tid; i < NBOX; i += 1024) {
    float sc = skey[(size_t)b * NBOX + i];
    unsigned k16 = okey(sc) >> 16;
    int pos = -1;
    if (k16 > sPivot16) {
      pos = (int)atomicAdd(&cnt0, 1u);
    } else if (k16 == sPivot16) {
      unsigned j = atomicAdd(&cnt1, 1u);
      if (j < sR) pos = (int)(sG + j);
    }
    if (pos >= 0) {
      csc[(size_t)b * PRE_NMS + pos] = sc;
      const float* bp = boxes + ((size_t)b * NBOX + i) * 4;
      float* cp = cbox + ((size_t)b * PRE_NMS + pos) * 4;
      cp[0] = bp[0]; cp[1] = bp[1]; cp[2] = bp[2]; cp[3] = bp[3];
    }
  }
}

// ===========================================================================
// greedy NMS: 300 x (block argmax over 6000 + IoU suppression). Scores in LDS.
// ===========================================================================
__global__ __launch_bounds__(1024)
void k_nms(const float* __restrict__ cbox, const float* __restrict__ csc,
           float* __restrict__ rois) {
  __shared__ float s[PRE_NMS];
  __shared__ float wv[32]; __shared__ int wi[32];
  __shared__ float bx[4];
  const int b = blockIdx.x, tid = threadIdx.x;
  const int lane = tid & 31, wid = tid >> 5;
  for (int j = tid; j < PRE_NMS; j += 1024) s[j] = csc[(size_t)b * PRE_NMS + j];
  __syncthreads();
  const float* CB = cbox + (size_t)b * PRE_NMS * 4;

  for (int it = 0; it < POST_NMS; ++it) {
    float bv = -3.0e38f; int bi = 0;
    for (int j = tid; j < PRE_NMS; j += 1024) {
      float v = s[j];
      if (v > bv) { bv = v; bi = j; }
    }
    #pragma unroll
    for (int off = 16; off; off >>= 1) {
      float ov = __shfl_down(bv, off); int oi = __shfl_down(bi, off);
      if (ov > bv) { bv = ov; bi = oi; }
    }
    if (lane == 0) { wv[wid] = bv; wi[wid] = bi; }
    __syncthreads();
    if (tid < 32) {
      bv = wv[tid]; bi = wi[tid];
      #pragma unroll
      for (int off = 16; off; off >>= 1) {
        float ov = __shfl_down(bv, off); int oi = __shfl_down(bi, off);
        if (ov > bv) { bv = ov; bi = oi; }
      }
      if (tid == 0) {
        float b0 = CB[(size_t)bi * 4], b1 = CB[(size_t)bi * 4 + 1];
        float b2 = CB[(size_t)bi * 4 + 2], b3 = CB[(size_t)bi * 4 + 3];
        bx[0] = b0; bx[1] = b1; bx[2] = b2; bx[3] = b3;
        float* R = rois + ((size_t)b * POST_NMS + it) * 4;
        R[0] = b0; R[1] = b1; R[2] = b2; R[3] = b3;
      }
    }
    __syncthreads();
    const float B0 = bx[0], B1 = bx[1], B2 = bx[2], B3 = bx[3];
    const float areaB = (B2 - B0) * (B3 - B1);
    for (int j = tid; j < PRE_NMS; j += 1024) {
      float q0 = CB[(size_t)j * 4], q1 = CB[(size_t)j * 4 + 1];
      float q2 = CB[(size_t)j * 4 + 2], q3 = CB[(size_t)j * 4 + 3];
      float ty  = fmaxf(B0, q0), tx  = fmaxf(B1, q1);
      float by2 = fminf(B2, q2), bx2 = fminf(B3, q3);
      float inter = fmaxf(by2 - ty, 0.f) * fmaxf(bx2 - tx, 0.f);
      float u = areaB + (q2 - q0) * (q3 - q1) - inter + 1e-9f;
      if (inter > 0.7f * u) s[j] = NEGV;   // selected box self-suppresses (IoU=1)
    }
    __syncthreads();
  }
}

__global__ void k_ridx(float* __restrict__ ridx) {
  int i = blockIdx.x * blockDim.x + threadIdx.x;
  if (i < BATCH * POST_NMS) ridx[i] = (float)(i / POST_NMS);
}

// ===========================================================================
extern "C" void kernel_launch(void* const* d_in, const int* in_sizes, int n_in,
                              void* d_out, int out_size, void* d_ws, size_t ws_size,
                              hipStream_t stream) {
  const float* x   = (const float*)d_in[0];
  const float* c1w = (const float*)d_in[1];
  const float* c1b = (const float*)d_in[2];
  const float* ow  = (const float*)d_in[3];
  const float* obb = (const float*)d_in[4];
  const float* lw  = (const float*)d_in[5];
  const float* lbb = (const float*)d_in[6];
  const int*   ihp = (const int*)d_in[7];
  const int*   iwp = (const int*)d_in[8];
  float* out = (float*)d_out;

  char* ws = (char*)d_ws;
  __bf16* xbf   = (__bf16*)(ws + OFF_XBF);
  __bf16* wfrag = (__bf16*)(ws + OFF_WFRAG);
  __bf16* feats = (__bf16*)(ws + OFF_FEATS);
  float*  fgws  = (float*)(ws + OFF_FG);
  float*  boxes = (float*)(ws + OFF_BOXES);
  float*  skey  = (float*)(ws + OFF_SKEY);
  float*  cbox  = (float*)(ws + OFF_CBOX);
  float*  csc   = (float*)(ws + OFF_CSC);

  float* out_obj  = out + O_OBJ;
  float* out_loc  = out + O_LOC;
  float* out_rois = out + O_ROIS;
  float* out_ridx = out + O_RIDX;
  float* out_anch = out + O_ANCH;

  const int NX = BATCH * CIN * HH * WW;        // 16,777,216
  const int NW = COUT * CIN * 9;               // 2,359,296

  k_cvt_x<<<(NX + 255) / 256, 256, 0, stream>>>(x, xbf, NX);
  k_reformat_w<<<(NW + 255) / 256, 256, 0, stream>>>(c1w, wfrag, NW);

  dim3 gconv(HH, COUT / 64, BATCH);            // (64, 8, 8)
  k_conv1<<<gconv, 256, 0, stream>>>(xbf, wfrag, c1b, feats);

  k_heads<<<BATCH * HH, 256, 0, stream>>>(feats, ow, obb, lw, lbb, out_obj, out_loc, fgws);
  k_anchors<<<(NBOX + 255) / 256, 256, 0, stream>>>(out_anch);
  k_decode<<<(BATCH * NBOX + 255) / 256, 256, 0, stream>>>(out_anch, out_loc, fgws,
                                                           ihp, iwp, boxes, skey);
  k_topk<<<BATCH, 1024, 0, stream>>>(skey, boxes, cbox, csc);
  k_nms<<<BATCH, 1024, 0, stream>>>(cbox, csc, out_rois);
  k_ridx<<<(BATCH * POST_NMS + 255) / 256, 256, 0, stream>>>(out_ridx);
}
